// ContactPositionRegressor_833223655563
// MI455X (gfx1250) — matrix operand order repaired and checked by reading, not compiled, not run on hardware
//
#include <hip/hip_runtime.h>
#include <hip/hip_bf16.h>

typedef __attribute__((ext_vector_type(16))) _Float16 v16h;
typedef __attribute__((ext_vector_type(8)))  float    v8f;
typedef __attribute__((ext_vector_type(4)))  unsigned u32x4;
typedef __attribute__((ext_vector_type(4)))  int      i32x4;
typedef __attribute__((ext_vector_type(8)))  int      i32x8;

#define NPTS  131072
#define SEGSZ 8192
#define BATCH 16
#define CH    512
#define HID   128     // C/4
#define EMBD  32
#define FUSED 1056
#define HH1   256
#define HH2   128
#define LN_EPS 1e-5f

// ---- ordered-uint encoding of float for atomicMax over signed floats ----
__device__ inline unsigned f2key(float f) {
  unsigned u = __float_as_uint(f);
  return (u & 0x80000000u) ? ~u : (u | 0x80000000u);
}
__device__ inline float key2f(unsigned k) {
  unsigned u = (k & 0x80000000u) ? (k ^ 0x80000000u) : ~k;
  return __uint_as_float(u);
}

__device__ inline void cvt4(v16h& a, int i, float4 q) {
  a[i+0] = (_Float16)q.x; a[i+1] = (_Float16)q.y;
  a[i+2] = (_Float16)q.z; a[i+3] = (_Float16)q.w;
}

// ---- Tensor Data Mover: 2D fp32 tile (rows x cols) global -> LDS ---------
// D# built per CDNA5 ISA 8.3/8.4 (group0: count/lds/global/type=2,
// group1: data_size=4B, tensor dims == tile dims, dim0 stride in elements).
__device__ inline void tdm_load_2d_f32(const float* gsrc, unsigned ldsByteAddr,
                                       int rows, int cols, int strideElems) {
  unsigned long long ga = (unsigned long long)(const void*)gsrc;
  u32x4 g0;
  g0[0] = 1u;                                            // count=1 (user mode)
  g0[1] = ldsByteAddr;                                   // lds_addr
  g0[2] = (unsigned)(ga & 0xFFFFFFFFull);                // global_addr[31:0]
  g0[3] = (unsigned)((ga >> 32) & 0x01FFFFFFull)         // global_addr[56:32]
        | (2u << 30);                                    // type=2 ("image")
  i32x8 g1;
  g1[0] = (int)(2u << 16);                               // data_size=4B
  g1[1] = (int)(((unsigned)cols & 0xFFFFu) << 16);       // tensor_dim0[15:0]
  g1[2] = (int)((((unsigned)cols >> 16) & 0xFFFFu)       // tensor_dim0[31:16]
        | (((unsigned)rows & 0xFFFFu) << 16));           // tensor_dim1[15:0]
  g1[3] = (int)((((unsigned)rows >> 16) & 0xFFFFu)       // tensor_dim1[31:16]
        | (((unsigned)cols & 0xFFFFu) << 16));           // tile_dim0
  g1[4] = (int)((unsigned)rows & 0xFFFFu);               // tile_dim1 (tile_dim2=0)
  g1[5] = strideElems;                                   // tensor_dim0_stride lo
  g1[6] = 0;                                             // stride hi / dim1_stride lo
  g1[7] = 0;
  i32x4 z4 = {0, 0, 0, 0};
#if __has_include(<hip/amd_detail/amd_gfx1250_TDM.h>)
  i32x8 z8 = {0, 0, 0, 0, 0, 0, 0, 0};
  __builtin_amdgcn_tensor_load_to_lds(g0, g1, z4, z4, z8, 0);
#else
  __builtin_amdgcn_tensor_load_to_lds(g0, g1, z4, z4, 0);
#endif
}

// ---------------------------------------------------------------- init ----
__global__ void init_kernel(unsigned* __restrict__ segmax,
                            float* __restrict__ segsum,
                            float* __restrict__ pooled) {
  int t = threadIdx.x;
  if (t < 2 * BATCH) { segmax[t] = 0u; segsum[t] = 0.0f; }
  for (int i = t; i < 2 * BATCH * CH; i += 256) pooled[i] = 0.0f;
}

// ------------------------------------------------- attention score GEMM ----
// scores = relu(feat @ w1 + b1) @ w2 + b2 ; also atomic segment max.
// block = 256 threads (8 waves), each wave owns 16 rows; workgroup = 128 rows.
__global__ __launch_bounds__(256) void score_kernel(
    const float* __restrict__ lf, const float* __restrict__ pf,
    const int* __restrict__ loff, const int* __restrict__ poff,
    const float* __restrict__ aw1, const float* __restrict__ ab1,
    const float* __restrict__ aw2, const float* __restrict__ ab2,
    const float* __restrict__ pw1, const float* __restrict__ pb1,
    const float* __restrict__ pw2, const float* __restrict__ pb2,
    float* __restrict__ scores, unsigned* __restrict__ segmax)
{
  __shared__ _Float16 ldsW[32768];    // 64KB: half of w1, swizzled B-fragment layout
  __shared__ float    ldsStage[8192]; // 32KB: TDM staging (64 rows x 128 cols fp32)

  const int f = blockIdx.y;
  const float* feat = f ? pf : lf;
  const float* w1   = f ? pw1 : aw1;
  const float* b1   = f ? pb1 : ab1;
  const float* w2   = f ? pw2 : aw2;
  const float  b2   = f ? pb2[0] : ab2[0];
  const int*   offs = f ? poff : loff;

  const int tid  = threadIdx.x;
  const int wave = tid >> 5;
  const int lane = tid & 31;
  const int r0w  = blockIdx.x * 128 + wave * 16;
  const int rowm = r0w + (lane & 15);
  const int hiK  = lane >> 4;                 // 0: K{0..7,16..23}, 1: K{8..15,24..31}
  const float* frow = feat + (long)rowm * CH;

  float b1v[8], w2v[8];
#pragma unroll
  for (int nt = 0; nt < 8; ++nt) {
    int n = nt * 16 + (lane & 15);
    b1v[nt] = b1[n]; w2v[nt] = w2[n];
  }

  v8f acc[8] = {};

  for (int phase = 0; phase < 2; ++phase) {
    // ---- stage 256 K-rows of w1 via the Tensor Data Mover, 64 rows/step ----
    for (int sub = 0; sub < 4; ++sub) {
      const float* chunk = w1 + (long)(phase * 256 + sub * 64) * HID;
      if (wave == 0) {
        tdm_load_2d_f32(chunk, (unsigned)(size_t)ldsStage, 64, HID, HID);
        __builtin_amdgcn_s_wait_tensorcnt(0);
      }
      __syncthreads();   // staging visible; also fences previous-phase compute
      // convert fp32 -> f16 and swizzle into B-fragment layout:
      //   lane = n%16 + (k & 8 ? 16 : 0), slot = (k & 7) | ((k & 16) >> 1)
      for (int idx = tid; idx < 8192; idx += 256) {
        int klo = idx >> 7;             // 0..63 within staged chunk
        int n   = idx & 127;
        int kl  = sub * 64 + klo;       // 0..255 within phase
        float v = ldsStage[klo * 128 + n];
        int kbl = kl >> 5, kk = kl & 31;
        int nt = n >> 4, nn = n & 15;
        ldsW[((kbl * 8 + nt) * 32 + (nn + ((kk & 8) << 1))) * 16 + ((kk & 7) | ((kk & 16) >> 1))] =
            (_Float16)v;
      }
      __syncthreads();   // staging consumed before next TDM overwrites it
    }

    if (phase == 0) __builtin_prefetch((const void*)(frow + 256), 0, 0);

#pragma unroll
    for (int kbl = 0; kbl < 8; ++kbl) {
      // A fragment: 16x32 f16, per-lane two runs of 8 consecutive K values
      int k0 = (phase * 8 + kbl) * 32 + hiK * 8;
      float4 q0 = *(const float4*)(frow + k0);
      float4 q1 = *(const float4*)(frow + k0 + 4);
      float4 q2 = *(const float4*)(frow + k0 + 16);
      float4 q3 = *(const float4*)(frow + k0 + 20);
      v16h a;
      cvt4(a, 0, q0); cvt4(a, 4, q1); cvt4(a, 8, q2); cvt4(a, 12, q3);

      // preload all 8 B fragments, then burst the WMMAs back-to-back
      v16h bf[8];
#pragma unroll
      for (int nt = 0; nt < 8; ++nt)
        bf[nt] = *reinterpret_cast<const v16h*>(
            &ldsW[((kbl * 8 + nt) * 32 + lane) * 16]);
#pragma unroll
      for (int nt = 0; nt < 8; ++nt)
        acc[nt] = __builtin_amdgcn_wmma_f32_16x16x32_f16(
            false, a, false, bf[nt], (short)0, acc[nt], false, false);
    }
  }

  // epilogue: per C-layout, acc[nt][j] = D[row j+8*hiK][col nt*16 + lane%16]
  float pj[8];
#pragma unroll
  for (int j = 0; j < 8; ++j) {
    float s = 0.0f;
#pragma unroll
    for (int nt = 0; nt < 8; ++nt) {
      float h = acc[nt][j] + b1v[nt];
      h = h > 0.0f ? h : 0.0f;
      s += h * w2v[nt];
    }
    // reduce across the 16 lanes of this half-wave (all columns of one row)
    s += __shfl_xor(s, 1, 16);
    s += __shfl_xor(s, 2, 16);
    s += __shfl_xor(s, 4, 16);
    s += __shfl_xor(s, 8, 16);
    pj[j] = s;
  }

  if ((lane & 15) == 0) {            // lanes 0 (rows j) and 16 (rows j+8)
    int rb = r0w + hiK * 8;
    long sbase = (long)f * NPTS;
    float mx = -3.4e38f;
#pragma unroll
    for (int j = 0; j < 8; ++j) {
      float s = pj[j] + b2;
      scores[sbase + rb + j] = s;
      mx = fmaxf(mx, s);
    }
    int R = blockIdx.x * 128;        // whole block lies in one segment
    int seg = 0;
    while (seg < BATCH - 1 && R >= offs[seg]) seg++;
    atomicMax(&segmax[f * BATCH + seg], f2key(mx));
  }
}

// ----------------------------------------------- exp + segment sum --------
__global__ __launch_bounds__(256) void expsum_kernel(
    const int* __restrict__ loff, const int* __restrict__ poff,
    float* __restrict__ scores, const unsigned* __restrict__ segmax,
    float* __restrict__ segsum)
{
  __shared__ float red[256];
  int tid = threadIdx.x;
  int gid = blockIdx.x * 256 + tid;
  int f = gid >> 17;                  // / NPTS
  int p = gid & (NPTS - 1);
  const int* offs = f ? poff : loff;
  int seg = 0;
  while (seg < BATCH - 1 && p >= offs[seg]) seg++;
  float m = key2f(segmax[f * BATCH + seg]);
  float e = __expf(scores[gid] - m);
  scores[gid] = e;
  red[tid] = e;
  __syncthreads();
  for (int s = 128; s > 0; s >>= 1) {
    if (tid < s) red[tid] += red[tid + s];
    __syncthreads();
  }
  if (tid == 0) atomicAdd(&segsum[f * BATCH + seg], red[0]);
}

// ------------------------------------------ weighted segment-sum pool -----
// grid 512: [f:2][seg:16][cchunk:2][pchunk:8]; thread = one channel.
__global__ __launch_bounds__(256) void pool_kernel(
    const float* __restrict__ lf, const float* __restrict__ pf,
    const float* __restrict__ scores, const float* __restrict__ segsum,
    float* __restrict__ pooled)
{
  int b = blockIdx.x;
  int f = b >> 8;
  int seg = (b >> 4) & 15;
  int cchunk = (b >> 3) & 1;
  int pchunk = b & 7;
  const float* feat = f ? pf : lf;
  int c = cchunk * 256 + threadIdx.x;
  float inv = 1.0f / segsum[f * BATCH + seg];
  long p0 = (long)seg * SEGSZ + (long)pchunk * 1024;
  const float* sc = scores + (long)f * NPTS;
  float acc = 0.0f;
  for (int i = 0; i < 1024; ++i) {
    long p = p0 + i;
    acc += sc[p] * feat[p * CH + c];
  }
  atomicAdd(&pooled[(long)(f * BATCH + seg) * CH + c], acc * inv);
}

// ------------------------------------------------------- MLP head ---------
__global__ __launch_bounds__(256) void head_kernel(
    const float* __restrict__ pooled, const int* __restrict__ cat,
    const float* __restrict__ cemb,
    const float* __restrict__ rw1, const float* __restrict__ rb1,
    const float* __restrict__ g1,  const float* __restrict__ be1,
    const float* __restrict__ rw2, const float* __restrict__ rb2,
    const float* __restrict__ g2,  const float* __restrict__ be2,
    const float* __restrict__ rw3, const float* __restrict__ rb3,
    float* __restrict__ out)
{
  __shared__ float G[16 * FUSED];     // fused input, later reused for activations
  __shared__ float Hs[16 * HH1];
  __shared__ float mu[16], rs[16];
  int tid = threadIdx.x;

  for (int idx = tid; idx < 16 * FUSED; idx += 256) {
    int bb = idx / FUSED, k = idx - bb * FUSED;
    float v;
    if (k < 512)       v = pooled[bb * 512 + k];
    else if (k < 1024) v = pooled[(16 + bb) * 512 + (k - 512)];
    else               v = cemb[cat[bb] * EMBD + (k - 1024)];
    G[idx] = v;
  }
  __syncthreads();

  for (int idx = tid; idx < 16 * HH1; idx += 256) {
    int bb = idx >> 8, o = idx & 255;
    float s = rb1[o];
    for (int k = 0; k < FUSED; ++k) s += G[bb * FUSED + k] * rw1[k * HH1 + o];
    Hs[idx] = s;
  }
  __syncthreads();
  if (tid < 16) {
    float s = 0.0f, s2 = 0.0f;
    for (int k = 0; k < HH1; ++k) { float x = Hs[tid * HH1 + k]; s += x; s2 += x * x; }
    float m = s / HH1, v = s2 / HH1 - m * m;
    mu[tid] = m; rs[tid] = rsqrtf(v + LN_EPS);
  }
  __syncthreads();
  for (int idx = tid; idx < 16 * HH1; idx += 256) {
    int bb = idx >> 8, o = idx & 255;
    float x = (Hs[idx] - mu[bb]) * rs[bb] * g1[o] + be1[o];
    G[idx] = x > 0.0f ? x : 0.0f;
  }
  __syncthreads();

  for (int idx = tid; idx < 16 * HH2; idx += 256) {
    int bb = idx >> 7, o = idx & 127;
    float s = rb2[o];
    for (int k = 0; k < HH1; ++k) s += G[bb * HH1 + k] * rw2[k * HH2 + o];
    Hs[bb * HH2 + o] = s;
  }
  __syncthreads();
  if (tid < 16) {
    float s = 0.0f, s2 = 0.0f;
    for (int k = 0; k < HH2; ++k) { float x = Hs[tid * HH2 + k]; s += x; s2 += x * x; }
    float m = s / HH2, v = s2 / HH2 - m * m;
    mu[tid] = m; rs[tid] = rsqrtf(v + LN_EPS);
  }
  __syncthreads();
  for (int idx = tid; idx < 16 * HH2; idx += 256) {
    int bb = idx >> 7, o = idx & 127;
    float x = (Hs[idx] - mu[bb]) * rs[bb] * g2[o] + be2[o];
    G[idx] = x > 0.0f ? x : 0.0f;
  }
  __syncthreads();

  for (int idx = tid; idx < 48; idx += 256) {
    int bb = idx / 3, o = idx - bb * 3;
    float s = rb3[o];
    for (int k = 0; k < HH2; ++k) s += G[bb * HH2 + k] * rw3[k * 3 + o];
    out[idx] = s;
  }
}

// ------------------------------------------------------------------------
extern "C" void kernel_launch(void* const* d_in, const int* in_sizes, int n_in,
                              void* d_out, int out_size, void* d_ws, size_t ws_size,
                              hipStream_t stream) {
  const float* lf   = (const float*)d_in[0];
  const float* pf   = (const float*)d_in[1];
  const int*   loff = (const int*)d_in[2];
  const int*   poff = (const int*)d_in[3];
  const int*   cat  = (const int*)d_in[4];
  const float* aw1  = (const float*)d_in[5];
  const float* ab1  = (const float*)d_in[6];
  const float* aw2  = (const float*)d_in[7];
  const float* ab2  = (const float*)d_in[8];
  const float* pw1  = (const float*)d_in[9];
  const float* pb1  = (const float*)d_in[10];
  const float* pw2  = (const float*)d_in[11];
  const float* pb2  = (const float*)d_in[12];
  const float* cemb = (const float*)d_in[13];
  const float* rw1  = (const float*)d_in[14];
  const float* rb1  = (const float*)d_in[15];
  const float* g1   = (const float*)d_in[16];
  const float* be1  = (const float*)d_in[17];
  const float* rw2  = (const float*)d_in[18];
  const float* rb2  = (const float*)d_in[19];
  const float* g2   = (const float*)d_in[20];
  const float* be2  = (const float*)d_in[21];
  const float* rw3  = (const float*)d_in[22];
  const float* rb3  = (const float*)d_in[23];
  float* out = (float*)d_out;

  float*    ws     = (float*)d_ws;
  float*    scores = ws;                            // 2*NPTS
  unsigned* segmax = (unsigned*)(ws + 2 * NPTS);    // 32
  float*    segsum = ws + 2 * NPTS + 32;            // 32
  float*    pooled = ws + 2 * NPTS + 64;            // 2*16*512

  init_kernel<<<1, 256, 0, stream>>>(segmax, segsum, pooled);
  score_kernel<<<dim3(NPTS / 128, 2), 256, 0, stream>>>(
      lf, pf, loff, poff, aw1, ab1, aw2, ab2, pw1, pb1, pw2, pb2, scores, segmax);
  expsum_kernel<<<(2 * NPTS) / 256, 256, 0, stream>>>(loff, poff, scores, segmax, segsum);
  pool_kernel<<<512, 256, 0, stream>>>(lf, pf, scores, segsum, pooled);
  head_kernel<<<1, 256, 0, stream>>>(pooled, cat, cemb, rw1, rb1, g1, be1,
                                     rw2, rb2, g2, be2, rw3, rb3, out);
}
